// FusionModule_42769284333645
// MI455X (gfx1250) — compile-verified, matrix operand order checked by reading, not run on hardware
//
#include <hip/hip_runtime.h>
#include <hip/hip_bf16.h>
#include <math.h>

// ---------------- problem constants ----------------
#define BATCH   8
#define CCH     256
#define HH      64
#define WW      64
#define HWN     4096            // H*W
#define HEADS   8
#define PTS     4
#define HD      32              // C / HEADS
#define CIN_G   1024
#define CIN_S   320
#define COUT    320

// ---------------- WMMA GEMM tiling ----------------
#define BDIM  256               // 8 waves (wave32)
#define TM    64
#define TN    128
#define TK    32
#define PITCH 80                // bytes per LDS row: 64B data + 16B pad (16B aligned, bank-spread)

typedef __attribute__((ext_vector_type(16))) __bf16 v16bf;
typedef __attribute__((ext_vector_type(8)))  __bf16 v8bf;
typedef __attribute__((ext_vector_type(8)))  float  v8f;

__device__ __forceinline__ __bf16 f2bf(float f) {
  unsigned int u = __builtin_bit_cast(unsigned int, f);
  unsigned int r = u + 0x7FFFu + ((u >> 16) & 1u);   // round-to-nearest-even
  unsigned short h = (unsigned short)(r >> 16);
  return __builtin_bit_cast(__bf16, h);
}

__device__ __forceinline__ v8bf pack8(float4 a, float4 b) {
  v8bf p;
  p[0] = f2bf(a.x); p[1] = f2bf(a.y); p[2] = f2bf(a.z); p[3] = f2bf(a.w);
  p[4] = f2bf(b.x); p[5] = f2bf(b.y); p[6] = f2bf(b.z); p[7] = f2bf(b.w);
  return p;
}

__device__ __forceinline__ v16bf frag_from_lds(const unsigned char* row, int half) {
  v8bf lo = *(const v8bf*)(row + half * 16);
  v8bf hi = *(const v8bf*)(row + 32 + half * 16);
  return __builtin_shufflevector(lo, hi, 0,1,2,3,4,5,6,7,8,9,10,11,12,13,14,15);
}

// -----------------------------------------------------------------------------
// Generic per-batch GEMM:  C[b] = act( A_eff[M,K] * B_eff[K,N] + bias )
//   A_eff = TRA ? A^T (A stored [K][M], lda = M-pitch) : A ([M][K], lda = K-pitch)
//   B_eff[k][n] = TRB ? B[n][k] (ldb=K-pitch)
//                     : B[k][n] (ldb=N-pitch)  (+ bscale * B1[k][n] if B1)
//   store: storeTrans ? C[n*ldc + m] : C[m*ldc + n]
// Block tile 64(M) x 128(N), K step 32; 8 waves: 4 M-strips x 2 N-halves,
// each wave computes four 16x16 D tiles reusing one A fragment.
// N multiple of 128 in all calls; K % 32 == 0; M % 8 == 0; 32B-aligned chunks.
// -----------------------------------------------------------------------------
template <int TRA, int TRB>
__global__ __launch_bounds__(BDIM)
void wmma_gemm_kernel(const float* __restrict__ A, const float* __restrict__ B0,
                      const float* __restrict__ B1, float bscale,
                      const float* __restrict__ bias, float* __restrict__ C,
                      int M, int K, int lda, int ldb, int ldc,
                      long sA, long sB, long sC,
                      int storeTrans, int doRelu)
{
  __shared__ __align__(16) unsigned char ldsA[TM * PITCH];
  __shared__ __align__(16) unsigned char ldsB[TN * PITCH];

  const int t  = threadIdx.x;
  const int bz = blockIdx.z;
  const int n0 = blockIdx.x * TN;
  const int m0 = blockIdx.y * TM;

  const float* Ab = A  + (size_t)bz * sA;
  const float* Bb = B0 + (size_t)bz * sB;
  float*       Cb = C  + (size_t)bz * sC;

  const int lane = t & 31;
  const int wave = t >> 5;
  const int rblk = wave >> 1;          // 0..3 : 16-row M strip
  const int cblk = wave & 1;           // 0..1 : 64-col N half
  const int lm   = lane & 15;
  const int half = lane >> 4;

  v8f d0 = {}; v8f d1 = {}; v8f d2 = {}; v8f d3 = {};

  const float4 fz = make_float4(0.f, 0.f, 0.f, 0.f);

  for (int kt = 0; kt < K; kt += TK) {
    // ---------------- stage A tile (64 x 32, bf16, M-major) ----------------
    if (!TRA) {
      const int r = t >> 2, q = t & 3;
      const int m = m0 + r;
      float4 f0 = fz, f1 = fz;
      if (m < M) {
        const float4* src = (const float4*)(Ab + (size_t)m * lda + kt + q * 8);
        f0 = src[0]; f1 = src[1];
      }
      *(v8bf*)(ldsA + r * PITCH + q * 16) = pack8(f0, f1);
    } else {
      const int kk = t >> 3, q = t & 7;
      float4 f0 = fz, f1 = fz;
      if (m0 + q * 8 < M) {            // exact: M % 8 == 0
        const float4* src = (const float4*)(Ab + (size_t)(kt + kk) * lda + m0 + q * 8);
        f0 = src[0]; f1 = src[1];
      }
      const float vals[8] = { f0.x, f0.y, f0.z, f0.w, f1.x, f1.y, f1.z, f1.w };
      #pragma unroll
      for (int i = 0; i < 8; ++i)
        ((__bf16*)(ldsA + (q * 8 + i) * PITCH))[kk] = f2bf(vals[i]);
    }
    // ---------------- stage B tile (128 x 32, bf16, N-major) ----------------
    if (!TRB) {
      const int kk = t >> 3, q = t & 7;
      #pragma unroll
      for (int h2 = 0; h2 < 2; ++h2) {
        const int nc = q * 8 + h2 * 64;
        const float* srcp = Bb + (size_t)(kt + kk) * ldb + n0 + nc;
        float4 f0 = ((const float4*)srcp)[0];
        float4 f1 = ((const float4*)srcp)[1];
        if (B1) {
          const float* s1p = B1 + (size_t)(kt + kk) * ldb + n0 + nc;
          const float4 g0 = ((const float4*)s1p)[0];
          const float4 g1 = ((const float4*)s1p)[1];
          f0.x += bscale * g0.x; f0.y += bscale * g0.y;
          f0.z += bscale * g0.z; f0.w += bscale * g0.w;
          f1.x += bscale * g1.x; f1.y += bscale * g1.y;
          f1.z += bscale * g1.z; f1.w += bscale * g1.w;
        }
        const float vals[8] = { f0.x, f0.y, f0.z, f0.w, f1.x, f1.y, f1.z, f1.w };
        #pragma unroll
        for (int i = 0; i < 8; ++i)
          ((__bf16*)(ldsB + (nc + i) * PITCH))[kk] = f2bf(vals[i]);
        if (kt + TK < K) __builtin_prefetch(srcp + (size_t)TK * ldb, 0, 1);
      }
    } else {
      const int r = t >> 2, q = t & 3;
      #pragma unroll
      for (int h2 = 0; h2 < 2; ++h2) {
        const int rr = r + h2 * 64;
        const float4* src = (const float4*)(Bb + (size_t)(n0 + rr) * ldb + kt + q * 8);
        *(v8bf*)(ldsB + rr * PITCH + q * 16) = pack8(src[0], src[1]);
        if (kt + TK < K) __builtin_prefetch((const float*)src + TK, 0, 1);
      }
    }
    __syncthreads();

    // ------------- fragments (ISA 16-bit 16x32 layout) + 4 WMMAs -------------
    // lane L: row = L%16, chunk lo = K[half*8 .. +8), chunk hi = K[16+half*8 .. +8)
    const v16bf afrag = frag_from_lds(ldsA + (rblk * 16 + lm) * PITCH, half);

    const unsigned char* bbase = ldsB + (cblk * 64 + lm) * PITCH;
    const v16bf bf0 = frag_from_lds(bbase,             half);
    const v16bf bf1 = frag_from_lds(bbase + 16 * PITCH, half);
    const v16bf bf2 = frag_from_lds(bbase + 32 * PITCH, half);
    const v16bf bf3 = frag_from_lds(bbase + 48 * PITCH, half);

    d0 = __builtin_amdgcn_wmma_f32_16x16x32_bf16(false, afrag, false, bf0,
                                                 (short)0, d0, false, false);
    d1 = __builtin_amdgcn_wmma_f32_16x16x32_bf16(false, afrag, false, bf1,
                                                 (short)0, d1, false, false);
    d2 = __builtin_amdgcn_wmma_f32_16x16x32_bf16(false, afrag, false, bf2,
                                                 (short)0, d2, false, false);
    d3 = __builtin_amdgcn_wmma_f32_16x16x32_bf16(false, afrag, false, bf3,
                                                 (short)0, d3, false, false);
    __syncthreads();
  }

  // --------------- epilogue: D layout lane L: n=L%16, m=r+8*half ---------------
  const int mrow = m0 + rblk * 16 + half * 8;
  const int nbase = n0 + cblk * 64 + lm;
  v8f dt[4] = { d0, d1, d2, d3 };
  #pragma unroll
  for (int r = 0; r < 8; ++r) {
    const int m = mrow + r;
    if (m < M) {
      const float bv = bias ? bias[m] : 0.f;
      #pragma unroll
      for (int j = 0; j < 4; ++j) {
        float v = dt[j][r] + bv;
        if (doRelu) v = fmaxf(v, 0.f);
        const int n = nbase + j * 16;
        if (!storeTrans) Cb[(size_t)m * ldc + n] = v;
        else             Cb[(size_t)n * ldc + m] = v;
      }
    }
  }
}

// -----------------------------------------------------------------------------
// LoFTR-style 2D sinusoidal PE: PE[c][n], c in [0,256), n = y*64 + x
// -----------------------------------------------------------------------------
__global__ void pe_kernel(float* __restrict__ pe)
{
  const int e = blockIdx.x * blockDim.x + threadIdx.x;
  if (e >= CCH * HWN) return;
  const int c = e >> 12;
  const int n = e & (HWN - 1);
  const int i = c >> 2;
  const int tt = c & 3;
  const float div = expf((float)(2 * i) * (-9.210340371976184f / 128.0f)); // -ln(1e4)/(C/2)
  const float xf = (float)(n & 63);
  const float yf = (float)(n >> 6);
  float v;
  if      (tt == 0) v = sinf(xf * div);
  else if (tt == 1) v = cosf(xf * div);
  else if (tt == 2) v = sinf(yf * div);
  else              v = cosf(yf * div);
  pe[e] = v;
}

// -----------------------------------------------------------------------------
// alpha[b] = mean_n sigmoid( sum_j h[b][j][n]*w2[j] + b2 ) ; also written to out tail
// -----------------------------------------------------------------------------
__global__ __launch_bounds__(256)
void alpha_kernel(const float* __restrict__ h, const float* __restrict__ w2,
                  const float* __restrict__ b2, float* __restrict__ alphaOut,
                  float* __restrict__ outTail)
{
  __shared__ float red[256];
  __shared__ float w2s[64];
  const int b = blockIdx.x;
  const int t = threadIdx.x;
  if (t < 64) w2s[t] = w2[t];
  __syncthreads();
  const float* hb = h + (size_t)b * 64 * HWN;
  const float bb = b2[0];
  float local = 0.f;
  for (int n = t; n < HWN; n += 256) {
    float z = bb;
    #pragma unroll 8
    for (int j = 0; j < 64; ++j) z += hb[(size_t)j * HWN + n] * w2s[j];
    local += 1.f / (1.f + expf(-z));
  }
  red[t] = local;
  __syncthreads();
  for (int s = 128; s > 0; s >>= 1) {
    if (t < s) red[t] += red[t + s];
    __syncthreads();
  }
  if (t == 0) {
    const float a = red[0] * (1.0f / (float)HWN);
    alphaOut[b] = a;
    outTail[b] = a;
  }
}

// -----------------------------------------------------------------------------
// Deformable sampling. One wave per (b, head, n); lane = channel d within head.
// V is [b][n][c] (c fastest -> coalesced gathers). attn written [b][n][c].
// -----------------------------------------------------------------------------
__global__ __launch_bounds__(256)
void sample_kernel(const float* __restrict__ V, const float* __restrict__ off,
                   const float* __restrict__ awl, float* __restrict__ attn)
{
  const int wave = (blockIdx.x << 3) + (threadIdx.x >> 5);
  const int d    = threadIdx.x & 31;
  const int n    = wave & (HWN - 1);
  const int hb   = wave >> 12;
  const int hd   = hb & (HEADS - 1);
  const int b    = hb >> 3;

  const float* offb = off + (size_t)b * (HEADS * PTS * 2) * HWN;
  const float* awlb = awl + (size_t)b * (HEADS * PTS) * HWN;
  const float* Vb   = V   + (size_t)b * HWN * CCH + hd * HD + d;
  float*       ab   = attn + (size_t)b * HWN * CCH;

  // softmax over the 4 points (scalar, redundantly computed per lane)
  float l0 = awlb[(size_t)(hd * PTS + 0) * HWN + n];
  float l1 = awlb[(size_t)(hd * PTS + 1) * HWN + n];
  float l2 = awlb[(size_t)(hd * PTS + 2) * HWN + n];
  float l3 = awlb[(size_t)(hd * PTS + 3) * HWN + n];
  float mx = fmaxf(fmaxf(l0, l1), fmaxf(l2, l3));
  float e0 = expf(l0 - mx), e1 = expf(l1 - mx), e2 = expf(l2 - mx), e3 = expf(l3 - mx);
  float inv = 1.f / (e0 + e1 + e2 + e3);
  float wp[PTS] = { e0 * inv, e1 * inv, e2 * inv, e3 * inv };

  const float rpx = (float)(n & 63) * (1.0f / 63.0f);
  const float rpy = (float)(n >> 6) * (1.0f / 63.0f);

  float acc = 0.f;
  #pragma unroll
  for (int p = 0; p < PTS; ++p) {
    const int j2 = (hd * PTS + p) * 2;
    const float ox = offb[(size_t)(j2 + 0) * HWN + n];
    const float oy = offb[(size_t)(j2 + 1) * HWN + n];
    const float px = (rpx + ox * (1.0f / (float)WW)) * (float)WW - 0.5f;
    const float py = (rpy + oy * (1.0f / (float)HH)) * (float)HH - 0.5f;
    const int x0 = (int)floorf(px);
    const int y0 = (int)floorf(py);
    const float wx = px - (float)x0;
    const float wy = py - (float)y0;
    float s = 0.f;
    #pragma unroll
    for (int cy = 0; cy < 2; ++cy) {
      #pragma unroll
      for (int cx = 0; cx < 2; ++cx) {
        const int xi = x0 + cx;
        const int yi = y0 + cy;
        const float w = (cx ? wx : 1.f - wx) * (cy ? wy : 1.f - wy);
        if (xi >= 0 && xi < WW && yi >= 0 && yi < HH)
          s += w * Vb[(size_t)(yi * WW + xi) * CCH];
      }
    }
    acc += wp[p] * s;
  }
  ab[(size_t)n * CCH + hd * HD + d] = acc;
}

// -----------------------------------------------------------------------------
// fused = a*(OPs + 2*(G+PE)) + (1-a)*(OPo + 2*(G+2PE))     (layout [b][c][n])
// -----------------------------------------------------------------------------
__global__ void combine_kernel(const float* __restrict__ ops, const float* __restrict__ opo,
                               const float* __restrict__ g, const float* __restrict__ pe,
                               const float* __restrict__ alpha, float* __restrict__ fused)
{
  const size_t e = (size_t)blockIdx.x * blockDim.x + threadIdx.x;
  if (e >= (size_t)BATCH * CCH * HWN) return;
  const int b = (int)(e >> 20);                 // C*HW = 2^20
  const size_t rem = e & ((1u << 20) - 1);
  const float a = alpha[b];
  const float pv = pe[rem];
  const float gv = g[e];
  const float so = ops[e] + 2.f * (gv + pv);
  const float oo = opo[e] + 2.f * (gv + 2.f * pv);
  fused[e] = a * so + (1.f - a) * oo;
}

// -----------------------------------------------------------------------------
extern "C" void kernel_launch(void* const* d_in, const int* in_sizes, int n_in,
                              void* d_out, int out_size, void* d_ws, size_t ws_size,
                              hipStream_t stream)
{
  (void)in_sizes; (void)n_in; (void)out_size; (void)ws_size;

  const float* ground   = (const float*)d_in[0];
  const float* sat      = (const float*)d_in[1];
  const float* osm      = (const float*)d_in[2];
  const float* gc_w     = (const float*)d_in[3];
  const float* gc_b     = (const float*)d_in[4];
  const float* sc_w     = (const float*)d_in[5];
  const float* sc_b     = (const float*)d_in[6];
  const float* up_w     = (const float*)d_in[7];
  const float* up_b     = (const float*)d_in[8];
  const float* a_w1     = (const float*)d_in[9];
  const float* a_b1     = (const float*)d_in[10];
  const float* a_w2     = (const float*)d_in[11];
  const float* a_b2     = (const float*)d_in[12];
  const float* s_vp_w   = (const float*)d_in[13];
  const float* s_vp_b   = (const float*)d_in[14];
  const float* s_so_w   = (const float*)d_in[15];
  const float* s_so_b   = (const float*)d_in[16];
  const float* s_aw_w   = (const float*)d_in[17];
  const float* s_aw_b   = (const float*)d_in[18];
  const float* s_op_w   = (const float*)d_in[19];
  const float* s_op_b   = (const float*)d_in[20];
  const float* o_vp_w   = (const float*)d_in[21];
  const float* o_vp_b   = (const float*)d_in[22];
  const float* o_so_w   = (const float*)d_in[23];
  const float* o_so_b   = (const float*)d_in[24];
  const float* o_aw_w   = (const float*)d_in[25];
  const float* o_aw_b   = (const float*)d_in[26];
  const float* o_op_w   = (const float*)d_in[27];
  const float* o_op_b   = (const float*)d_in[28];
  // d_in[29] = index (unused, params for index 0 pre-selected)

  float* out = (float*)d_out;

  // ---- workspace layout (floats); with reuse ≈ 240 MB ----
  float* ws = (float*)d_ws;
  size_t off = 0;
  const size_t SZ_CHW  = (size_t)BATCH * CCH * HWN;          // 8,388,608
  const size_t SZ_PE   = (size_t)CCH * HWN;                  // 1,048,576
  const size_t SZ_OFF  = (size_t)BATCH * (HEADS * PTS * 2) * HWN; // 2,097,152
  const size_t SZ_AWL  = (size_t)BATCH * (HEADS * PTS) * HWN;     // 1,048,576
  const size_t SZ_H    = (size_t)BATCH * 64 * HWN;                // 2,097,152

  float* PE    = ws + off; off += SZ_PE;
  float* G     = ws + off; off += SZ_CHW;
  float* S     = ws + off; off += SZ_CHW;   // later reused as attnS
  float* O     = ws + off; off += SZ_CHW;   // later reused as attnO
  float* Vs    = ws + off; off += SZ_CHW;   // later reused as OPs
  float* Vo    = ws + off; off += SZ_CHW;   // later reused as OPo
  float* offS  = ws + off; off += SZ_OFF;
  float* offO  = ws + off; off += SZ_OFF;
  float* awlS  = ws + off; off += SZ_AWL;
  float* awlO  = ws + off; off += SZ_AWL;
  float* hbuf  = ws + off; off += SZ_H;
  float* fused = ws + off; off += SZ_CHW;
  float* alphaBuf = ws + off; off += 8;

  float* attnS = S;   // S dead after V_sat GEMM consumes it
  float* attnO = O;
  float* OPs   = Vs;  // Vs dead after sample_sat
  float* OPo   = Vo;

  const long sCHW  = (long)CCH * HWN;
  const long sGRD  = (long)CIN_G * HWN;
  const long sSAT  = (long)CIN_S * HWN;
  const long sNC   = (long)HWN * CCH;      // [n][c] layouts
  const long sOFF  = (long)(HEADS * PTS * 2) * HWN;
  const long sAWL  = (long)(HEADS * PTS) * HWN;
  const long sH    = (long)64 * HWN;
  const long sOUT  = (long)COUT * HWN;

  const dim3 blk(BDIM);
  const int NT = HWN / TN;                 // 32 N-tiles

  // 0) position encoding
  pe_kernel<<<(int)((SZ_PE + 255) / 256), 256, 0, stream>>>(PE);

  // 1) G = gc_w[256,1024] x ground + gc_b
  wmma_gemm_kernel<0,0><<<dim3(NT, 4, BATCH), blk, 0, stream>>>(
      gc_w, ground, nullptr, 0.f, gc_b, G,
      CCH, CIN_G, CIN_G, HWN, HWN, 0, sGRD, sCHW, 0, 0);

  // 2) S = sc_w x sat ;  3) O = sc_w x osm   (source bug preserved: osm uses sat_convs)
  wmma_gemm_kernel<0,0><<<dim3(NT, 4, BATCH), blk, 0, stream>>>(
      sc_w, sat, nullptr, 0.f, sc_b, S,
      CCH, CIN_S, CIN_S, HWN, HWN, 0, sSAT, sCHW, 0, 0);
  wmma_gemm_kernel<0,0><<<dim3(NT, 4, BATCH), blk, 0, stream>>>(
      sc_w, osm, nullptr, 0.f, sc_b, O,
      CCH, CIN_S, CIN_S, HWN, HWN, 0, sSAT, sCHW, 0, 0);

  // 4) h = relu(a_w1^T x G + a_b1)   (a_w1 stored [256][64] -> transA)
  wmma_gemm_kernel<1,0><<<dim3(NT, 1, BATCH), blk, 0, stream>>>(
      a_w1, G, nullptr, 0.f, a_b1, hbuf,
      64, CCH, 64, HWN, HWN, 0, sCHW, sH, 0, 1);

  // 5) alpha (also writes second output at tail of d_out)
  alpha_kernel<<<BATCH, 256, 0, stream>>>(hbuf, a_w2, a_b2, alphaBuf,
                                          out + (size_t)BATCH * COUT * HWN);

  // 6) V_sat = vp_w^T x S  -> stored [n][c] ; V_osm likewise
  wmma_gemm_kernel<1,0><<<dim3(NT, 4, BATCH), blk, 0, stream>>>(
      s_vp_w, S, nullptr, 0.f, s_vp_b, Vs,
      CCH, CCH, CCH, HWN, CCH, 0, sCHW, sNC, 1, 0);
  wmma_gemm_kernel<1,0><<<dim3(NT, 4, BATCH), blk, 0, stream>>>(
      o_vp_w, O, nullptr, 0.f, o_vp_b, Vo,
      CCH, CCH, CCH, HWN, CCH, 0, sCHW, sNC, 1, 0);

  // 7) offsets / attention-weight logits ; query = G + PE (sat) / G + 2*PE (osm)
  wmma_gemm_kernel<1,0><<<dim3(NT, 1, BATCH), blk, 0, stream>>>(
      s_so_w, G, PE, 1.f, s_so_b, offS,
      HEADS * PTS * 2, CCH, HEADS * PTS * 2, HWN, HWN, 0, sCHW, sOFF, 0, 0);
  wmma_gemm_kernel<1,0><<<dim3(NT, 1, BATCH), blk, 0, stream>>>(
      s_aw_w, G, PE, 1.f, s_aw_b, awlS,
      HEADS * PTS, CCH, HEADS * PTS, HWN, HWN, 0, sCHW, sAWL, 0, 0);
  wmma_gemm_kernel<1,0><<<dim3(NT, 1, BATCH), blk, 0, stream>>>(
      o_so_w, G, PE, 2.f, o_so_b, offO,
      HEADS * PTS * 2, CCH, HEADS * PTS * 2, HWN, HWN, 0, sCHW, sOFF, 0, 0);
  wmma_gemm_kernel<1,0><<<dim3(NT, 1, BATCH), blk, 0, stream>>>(
      o_aw_w, G, PE, 2.f, o_aw_b, awlO,
      HEADS * PTS, CCH, HEADS * PTS, HWN, HWN, 0, sCHW, sAWL, 0, 0);

  // 8) deformable sampling: one wave per (b, head, n)
  const int sampleBlocks = (BATCH * HEADS * HWN) / 8;   // 8 waves per block
  sample_kernel<<<sampleBlocks, 256, 0, stream>>>(Vs, offS, awlS, attnS);
  sample_kernel<<<sampleBlocks, 256, 0, stream>>>(Vo, offO, awlO, attnO);

  // 9) output projections: OP = op_w^T x attn ([n][c] input -> transB)
  wmma_gemm_kernel<1,1><<<dim3(NT, 4, BATCH), blk, 0, stream>>>(
      s_op_w, attnS, nullptr, 0.f, s_op_b, OPs,
      CCH, CCH, CCH, CCH, HWN, 0, sNC, sCHW, 0, 0);
  wmma_gemm_kernel<1,1><<<dim3(NT, 4, BATCH), blk, 0, stream>>>(
      o_op_w, attnO, nullptr, 0.f, o_op_b, OPo,
      CCH, CCH, CCH, CCH, HWN, 0, sNC, sCHW, 0, 0);

  // 10) fused = a*(OPs + 2*(G+PE)) + (1-a)*(OPo + 2*(G+2PE))
  combine_kernel<<<(int)((SZ_CHW + 255) / 256), 256, 0, stream>>>(
      OPs, OPo, G, PE, alphaBuf, fused);

  // 11) out = up_w[320,256] x fused + up_b   -> d_out [B,320,64,64]
  wmma_gemm_kernel<0,0><<<dim3(NT, 5, BATCH), blk, 0, stream>>>(
      up_w, fused, nullptr, 0.f, up_b, out,
      COUT, CCH, CCH, HWN, HWN, 0, sCHW, sOUT, 0, 0);
}
